// GarmentNet_LF_18253611008151
// MI455X (gfx1250) — matrix operand order repaired and checked by reading, not compiled, not run on hardware
//
#include <hip/hip_runtime.h>
#include <hip/hip_bf16.h>

typedef __attribute__((ext_vector_type(2)))  float     v2f;
typedef __attribute__((ext_vector_type(8)))  float     v8f;
typedef __attribute__((ext_vector_type(8)))  _Float16  v8h;
typedef __attribute__((ext_vector_type(16))) _Float16  v16h;

#define EPS 1e-5f

// ---------------------------------------------------------------------------
// MNet GEMM: out[M,N] = A[M,K] @ W[N,K]^T + bias[N]  (all f32, WMMA f32 16x16x4)
// One wave per 16x16 tile, 8 waves / block.
// ---------------------------------------------------------------------------
__global__ void mnet_gemm_f32(const float* __restrict__ A,
                              const float* __restrict__ W,
                              const float* __restrict__ bias,
                              float* __restrict__ out,
                              int M, int N, int K) {
  const int wave = threadIdx.x >> 5;
  const int lane = threadIdx.x & 31;
  const int tile = blockIdx.x * 8 + wave;
  const int tilesN = N >> 4;
  const int tm = (tile / tilesN) << 4;
  const int tn = (tile % tilesN) << 4;
  const int hi = lane >> 4;     // 0 or 1
  const int lm = lane & 15;

  const float* arow = A + (size_t)(tm + lm) * K + 2 * hi;  // A[m, k+2hi .. +1]
  const float* wrow = W + (size_t)(tn + lm) * K + 2 * hi;  // B[k][n] = W[n][k]
  v8f c = {};
  for (int k = 0; k < K; k += 4) {
    v2f a = *(const v2f*)(arow + k);
    v2f b = *(const v2f*)(wrow + k);
    c = __builtin_amdgcn_wmma_f32_16x16x4_f32(false, a, false, b, (short)0, c,
                                              false, false);
  }
  const float bv = bias[tn + lm];
  float* o = out + (size_t)(tm + 8 * hi) * N + tn + lm;  // C: m = r + 8*hi, n = lm
#pragma unroll
  for (int r = 0; r < 8; ++r) o[(size_t)r * N] = c[r] + bv;
}

// ---------------------------------------------------------------------------
// Fused training-mode BatchNorm (biased var over M rows) + PReLU(scalar alpha).
// Block = 32 cols x 8 row-groups. OutT = float (MNet) or _Float16 (expert path).
// ---------------------------------------------------------------------------
template <typename OutT>
__global__ void bn_prelu(const float* __restrict__ X, OutT* __restrict__ Y,
                         const float* __restrict__ g, const float* __restrict__ b,
                         const float* __restrict__ alpha, int M, int N) {
  __shared__ float ssum[8][32];
  __shared__ float ssq[8][32];
  const int cl = threadIdx.x & 31;
  const int rg = threadIdx.x >> 5;
  const int col = blockIdx.x * 32 + cl;

  float s = 0.f, q = 0.f;
  for (int r = rg; r < M; r += 8) {
    float v = X[(size_t)r * N + col];
    s += v; q += v * v;
  }
  ssum[rg][cl] = s; ssq[rg][cl] = q;
  __syncthreads();
  if (rg == 0) {
    float ts = 0.f, tq = 0.f;
#pragma unroll
    for (int i = 0; i < 8; ++i) { ts += ssum[i][cl]; tq += ssq[i][cl]; }
    float mu = ts / (float)M;
    float var = tq / (float)M - mu * mu;
    ssum[0][cl] = mu;
    ssq[0][cl] = rsqrtf(var + EPS);
  }
  __syncthreads();
  const float mu = ssum[0][cl];
  const float is = ssq[0][cl];
  const float gg = g[col], bb = b[col], al = alpha[0];
  for (int r = rg; r < M; r += 8) {
    float v = (X[(size_t)r * N + col] - mu) * is * gg + bb;
    v = (v >= 0.f) ? v : al * v;
    Y[(size_t)r * N + col] = (OutT)v;
  }
}

// ---------------------------------------------------------------------------
// Full-tensor sum (single block) and wn = w / wsum.
// ---------------------------------------------------------------------------
__global__ void total_sum(const float* __restrict__ X, float* __restrict__ out,
                          int n) {
  __shared__ float sh[256];
  float s = 0.f;
  for (int i = threadIdx.x; i < n; i += 256) s += X[i];
  sh[threadIdx.x] = s;
  __syncthreads();
  for (int off = 128; off > 0; off >>= 1) {
    if ((int)threadIdx.x < off) sh[threadIdx.x] += sh[threadIdx.x + off];
    __syncthreads();
  }
  if (threadIdx.x == 0) out[0] = sh[0];
}

__global__ void scale_by_invsum(const float* __restrict__ w,
                                const float* __restrict__ wsum,
                                float* __restrict__ wn, int n) {
  int i = blockIdx.x * 256 + threadIdx.x;
  if (i < n) wn[i] = w[i] * (1.0f / wsum[0]);
}

__global__ void cvt_to_f16(const float* __restrict__ x, _Float16* __restrict__ y,
                           int n) {
  int i = blockIdx.x * 256 + threadIdx.x;
  if (i < n) y[i] = (_Float16)x[i];
}

// ---------------------------------------------------------------------------
// Expert mixture layer: out[M,O] = Sum_e wn[:,e] * (Z @ W_e^T) + wn @ bias
//   = (wn (x) Z)[M, E*I] @ Wr[E*I, O] + wn[M,E] @ bias[E,O]
// The wn[b,e] scale is applied to the f16 A fragment (per-lane packed mul).
// Z fragments for the whole I-dimension are preloaded into registers once
// (I/32 * 16 halves per lane) and reused across all 64 experts, so the steady
// state loop issues only the weight-fragment loads (32B/lane) per WMMA.
// Bias term initializes the SAME v8f accumulator via f32 WMMA (same C layout).
// ---------------------------------------------------------------------------
template <int I>
__global__ void expert_gemm(const _Float16* __restrict__ Z,
                            const _Float16* __restrict__ W,
                            const float* __restrict__ bias,
                            const float* __restrict__ wn,
                            float* __restrict__ out, int M, int O) {
  const int E = 64;
  const int NC = I / 32;  // K-chunks per expert
  const int wave = threadIdx.x >> 5;
  const int lane = threadIdx.x & 31;
  const int tile = blockIdx.x * 8 + wave;
  const int tilesN = O >> 4;
  const int tm = (tile / tilesN) << 4;
  const int tn = (tile % tilesN) << 4;
  const int hi = lane >> 4;
  const int lm = lane & 15;

  v8f c = {};

  // ---- bias term: wn[M,E] @ bias[E,O] via f32 WMMA 16x16x4 ----
  const float* wnrow = wn + (size_t)(tm + lm) * E + 2 * hi;
#pragma unroll
  for (int e = 0; e < E; e += 4) {
    v2f a = *(const v2f*)(wnrow + e);
    v2f b;
    b.x = bias[(size_t)(e + 2 * hi) * O + tn + lm];
    b.y = bias[(size_t)(e + 2 * hi + 1) * O + tn + lm];
    c = __builtin_amdgcn_wmma_f32_16x16x4_f32(false, a, false, b, (short)0, c,
                                              false, false);
  }

  // ---- preload all Z fragments of this 16-row tile (unscaled) ----
  // A fragment layout (f16 16x32): lane holds K = [k0+8hi .. +7] (VGPR0-3)
  // and K = [k0+16+8hi .. +7] (VGPR4-7) of row m = lm.
  const _Float16* zrow = Z + (size_t)(tm + lm) * I;
  v16h zf[NC];
#pragma unroll
  for (int ch = 0; ch < NC; ++ch) {
    v8h a0 = *(const v8h*)(zrow + 32 * ch + 8 * hi);
    v8h a1 = *(const v8h*)(zrow + 32 * ch + 16 + 8 * hi);
#pragma unroll
    for (int t = 0; t < 8; ++t) { zf[ch][t] = a0[t]; zf[ch][8 + t] = a1[t]; }
  }

  // ---- main term: Sum_e wn[m,e] * Z[m,:] @ W_e^T, f16 WMMA K=32 ----
  const float* wnr = wn + (size_t)(tm + lm) * E;
  for (int e = 0; e < E; ++e) {
    const _Float16 s = (_Float16)wnr[e];
    const _Float16* we = W + ((size_t)e * O + tn + lm) * I;  // row n=lm, expert e
    if (e + 1 < E)  // pull next expert's weight row toward L1 (global_prefetch_b8)
      __builtin_prefetch(W + ((size_t)(e + 1) * O + tn + lm) * I, 0, 0);
#pragma unroll
    for (int ch = 0; ch < NC; ++ch) {
      v16h a = zf[ch] * s;  // v_pk_mul_f16 x8
      // B fragment: lane holds K = [32*ch+8hi .. +7] and [32*ch+16+8hi .. +7], col n=lm
      v8h b0 = *(const v8h*)(we + 32 * ch + 8 * hi);
      v8h b1 = *(const v8h*)(we + 32 * ch + 16 + 8 * hi);
      v16h bfr;
#pragma unroll
      for (int t = 0; t < 8; ++t) { bfr[t] = b0[t]; bfr[8 + t] = b1[t]; }
      c = __builtin_amdgcn_wmma_f32_16x16x32_f16(false, a, false, bfr, (short)0,
                                                 c, false, false);
    }
  }

  float* o = out + (size_t)(tm + 8 * hi) * O + tn + lm;
#pragma unroll
  for (int r = 0; r < 8; ++r) o[(size_t)r * O] = c[r];
}

// ---------------------------------------------------------------------------
extern "C" void kernel_launch(void* const* d_in, const int* in_sizes, int n_in,
                              void* d_out, int out_size, void* d_ws,
                              size_t ws_size, hipStream_t stream) {
  const int B = 4096, E = 64;
  const float* m0  = (const float*)d_in[0];
  const float* x0  = (const float*)d_in[1];
  const float* mW1 = (const float*)d_in[2];  const float* mb1 = (const float*)d_in[3];
  const float* mg1 = (const float*)d_in[4];  const float* mbe1= (const float*)d_in[5];
  const float* ma1 = (const float*)d_in[6];
  const float* mW2 = (const float*)d_in[7];  const float* mb2 = (const float*)d_in[8];
  const float* mg2 = (const float*)d_in[9];  const float* mbe2= (const float*)d_in[10];
  const float* ma2 = (const float*)d_in[11];
  const float* mW3 = (const float*)d_in[12]; const float* mb3 = (const float*)d_in[13];
  const float* mg3 = (const float*)d_in[14]; const float* mbe3= (const float*)d_in[15];
  const float* ma3 = (const float*)d_in[16];
  const float* Wenc0 = (const float*)d_in[17]; const float* benc0 = (const float*)d_in[18];
  const float* Wenc1 = (const float*)d_in[19]; const float* benc1 = (const float*)d_in[20];
  const float* Wdec0 = (const float*)d_in[21]; const float* bdec0 = (const float*)d_in[22];
  const float* Wdec1 = (const float*)d_in[23]; const float* bdec1 = (const float*)d_in[24];
  const float* bng = (const float*)d_in[25]; const float* bnb = (const float*)d_in[26];
  const float* aP  = (const float*)d_in[27];
  float* out = (float*)d_out;

  // ---- workspace layout (256B aligned regions) ----
  char* ws = (char*)d_ws;
  size_t off = 0;
  auto alloc = [&](size_t bytes) { void* p = ws + off; off += (bytes + 255) & ~(size_t)255; return p; };
  float*    buf0   = (float*)   alloc((size_t)B * 256 * 4);  // pre-activation
  float*    buf1   = (float*)   alloc((size_t)B * 256 * 4);  // MNet activations
  float*    wbuf   = (float*)   alloc((size_t)B * E * 4);    // w (post MNet)
  float*    wnbuf  = (float*)   alloc((size_t)B * E * 4);    // wn = w / sum(w)
  float*    wsum   = (float*)   alloc(256);
  _Float16* act16  = (_Float16*)alloc((size_t)B * 256 * 2);  // f16 activations
  _Float16* w16e0  = (_Float16*)alloc((size_t)E * 256 * 64 * 2);
  _Float16* w16e1  = (_Float16*)alloc((size_t)E * 256 * 256 * 2);
  _Float16* w16d0  = (_Float16*)alloc((size_t)E * 256 * 256 * 2);
  _Float16* w16d1  = (_Float16*)alloc((size_t)E * 64 * 256 * 2);
  (void)ws_size; (void)n_in; (void)in_sizes; (void)out_size;

  // ---- MNet gating MLP (f32 WMMA) ----
  mnet_gemm_f32<<<(B/16)*(256/16)/8, 256, 0, stream>>>(m0,  mW1, mb1, buf0, B, 256, 128);
  bn_prelu<float><<<256/32, 256, 0, stream>>>(buf0, buf1, mg1, mbe1, ma1, B, 256);
  mnet_gemm_f32<<<(B/16)*(128/16)/8, 256, 0, stream>>>(buf1, mW2, mb2, buf0, B, 128, 256);
  bn_prelu<float><<<128/32, 256, 0, stream>>>(buf0, buf1, mg2, mbe2, ma2, B, 128);
  mnet_gemm_f32<<<(B/16)*(64/16)/8, 256, 0, stream>>>(buf1, mW3, mb3, buf0, B, 64, 128);
  bn_prelu<float><<<64/32, 256, 0, stream>>>(buf0, wbuf, mg3, mbe3, ma3, B, 64);

  // ---- wn = w / sum(w) ----
  total_sum<<<1, 256, 0, stream>>>(wbuf, wsum, B * E);
  scale_by_invsum<<<(B * E + 255) / 256, 256, 0, stream>>>(wbuf, wsum, wnbuf, B * E);

  // ---- one-time f16 conversion of x0 and expert weights ----
  cvt_to_f16<<<(B * 64 + 255) / 256, 256, 0, stream>>>(x0, act16, B * 64);
  cvt_to_f16<<<(E*256*64  + 255) / 256, 256, 0, stream>>>(Wenc0, w16e0, E*256*64);
  cvt_to_f16<<<(E*256*256 + 255) / 256, 256, 0, stream>>>(Wenc1, w16e1, E*256*256);
  cvt_to_f16<<<(E*256*256 + 255) / 256, 256, 0, stream>>>(Wdec0, w16d0, E*256*256);
  cvt_to_f16<<<(E*64*256  + 255) / 256, 256, 0, stream>>>(Wdec1, w16d1, E*64*256);

  // ---- expert mixture layers (f16 WMMA, f32 accumulate) ----
  expert_gemm<64> <<<(B/16)*(256/16)/8, 256, 0, stream>>>(act16, w16e0, benc0, wnbuf, buf0, B, 256);
  bn_prelu<_Float16><<<256/32, 256, 0, stream>>>(buf0, act16, bng, bnb, aP, B, 256);
  expert_gemm<256><<<(B/16)*(256/16)/8, 256, 0, stream>>>(act16, w16e1, benc1, wnbuf, buf0, B, 256);
  bn_prelu<_Float16><<<256/32, 256, 0, stream>>>(buf0, act16, bng, bnb, aP, B, 256);
  expert_gemm<256><<<(B/16)*(256/16)/8, 256, 0, stream>>>(act16, w16d0, bdec0, wnbuf, buf0, B, 256);
  bn_prelu<_Float16><<<256/32, 256, 0, stream>>>(buf0, act16, bng, bnb, aP, B, 256);
  expert_gemm<256><<<(B/16)*(64/16)/8, 256, 0, stream>>>(act16, w16d1, bdec1, wnbuf, out, B, 64);
}
